// DCN_53867479826660
// MI455X (gfx1250) — matrix operand order repaired and checked by reading, not compiled, not run on hardware
//
#include <hip/hip_runtime.h>

typedef __attribute__((ext_vector_type(16))) __bf16 v16bf;
typedef __attribute__((ext_vector_type(2)))  __bf16 v2bf;
typedef __attribute__((ext_vector_type(8)))  float  v8f;

#define CHI  64
#define CHO  64
#define KK   9
#define Bn   4
#define Hh   128
#define Ww   256
#define HW   (Hh * Ww)        // 32768 = 2^15
#define NPIX (Bn * HW)        // 131072

// ---------------------------------------------------------------------------
// Kernel A: offset/mask 3x3 conv (64 -> 27 ch), then fully precompute the
// bilinear sampling state per (tap, pixel):
//   aT = a00 | (a01<<16), aB = a10 | (a11<<16)   (clamped in-bounds addresses)
//   w00..w11 = mask-premultiplied corner weights (0 for invalid corners)
// This hoists all floor/fract/clamp/validity math out of the channel loop in
// kernel B (it is shared across all 64 channels -> 64x redundancy otherwise).
// ---------------------------------------------------------------------------
__global__ __launch_bounds__(256)
void dcn_offset_kernel(const float* __restrict__ x,
                       const float* __restrict__ w_off,
                       const float* __restrict__ b_off,
                       unsigned int* __restrict__ aT_out,
                       unsigned int* __restrict__ aB_out,
                       float* __restrict__ w00_out,
                       float* __restrict__ w01_out,
                       float* __restrict__ w10_out,
                       float* __restrict__ w11_out) {
    const int p = blockIdx.x * 256 + threadIdx.x;
    const int b = p >> 15;
    const int r = p & (HW - 1);
    const int h = r / Ww;
    const int w = r - h * Ww;

    float acc[27];
#pragma unroll
    for (int o = 0; o < 27; ++o) acc[o] = b_off[o];

    const float* xb = x + (size_t)b * CHI * HW;
    for (int c = 0; c < CHI; ++c) {
        const float* xc = xb + c * HW;
        float xr[9];
#pragma unroll
        for (int ky = 0; ky < 3; ++ky) {
            const int iy = h - 1 + ky;
            const bool vy = (iy >= 0) & (iy < Hh);
#pragma unroll
            for (int kx = 0; kx < 3; ++kx) {
                const int ix = w - 1 + kx;
                const bool vv = vy & (ix >= 0) & (ix < Ww);
                xr[ky * 3 + kx] = vv ? xc[iy * Ww + ix] : 0.0f;
            }
        }
#pragma unroll
        for (int o = 0; o < 27; ++o) {
            const float* wo = w_off + (o * CHI + c) * KK;  // uniform -> s_load
#pragma unroll
            for (int t = 0; t < 9; ++t) acc[o] = fmaf(xr[t], wo[t], acc[o]);
        }
    }

#pragma unroll
    for (int k = 0; k < KK; ++k) {
        const float sy = (float)(h - 1 + k / 3) + acc[2 * k];
        const float sx = (float)(w - 1 + k % 3) + acc[2 * k + 1];
        const float mv = 1.0f / (1.0f + __expf(-acc[18 + k]));

        const float y0f = floorf(sy), x0f = floorf(sx);
        const float dy = sy - y0f, dx = sx - x0f;
        const int y0 = (int)y0f, x0 = (int)x0f;
        const int y1 = y0 + 1, x1 = x0 + 1;

        const int y0c = min(max(y0, 0), Hh - 1);
        const int y1c = min(max(y1, 0), Hh - 1);
        const int x0c = min(max(x0, 0), Ww - 1);
        const int x1c = min(max(x1, 0), Ww - 1);

        const float wy0 = ((y0 >= 0) & (y0 < Hh)) ? (1.0f - dy) : 0.0f;
        const float wy1 = ((y1 >= 0) & (y1 < Hh)) ? dy : 0.0f;
        const float wx0 = ((x0 >= 0) & (x0 < Ww)) ? (1.0f - dx) : 0.0f;
        const float wx1 = ((x1 >= 0) & (x1 < Ww)) ? dx : 0.0f;

        const unsigned int a00 = (unsigned int)(y0c * Ww + x0c);
        const unsigned int a01 = (unsigned int)(y0c * Ww + x1c);
        const unsigned int a10 = (unsigned int)(y1c * Ww + x0c);
        const unsigned int a11 = (unsigned int)(y1c * Ww + x1c);

        const int idx = k * NPIX + p;
        aT_out[idx]  = a00 | (a01 << 16);
        aB_out[idx]  = a10 | (a11 << 16);
        w00_out[idx] = mv * wy0 * wx0;
        w01_out[idx] = mv * wy0 * wx1;
        w10_out[idx] = mv * wy1 * wx0;
        w11_out[idx] = mv * wy1 * wx1;
    }
}

// ---------------------------------------------------------------------------
// Kernel B: fused gather + main GEMM on v_wmma_f32_16x16x32_bf16.
// WG = 256 thr (8 waves), tile = 64 out-ch x 64 pixels; wave grid 4(m) x 2(n),
// each wave owns a 16x32 C tile (two v8f accumulators).
// Per tap: stage weights (A) and gathered activations (B) into LDS in WMMA
// fragment order (each lane fragment = contiguous 32B -> ds_load_b128 pairs).
// Gather staging is now pure address-unpack + 8 loads + 8 FMA per channel
// pair; adjacent channels share address VGPRs via the +HW immediate offset
// (all 64 pixels of a block are in the same batch image).
// ---------------------------------------------------------------------------
__global__ __launch_bounds__(256)
void dcn_main_kernel(const float* __restrict__ x,
                     const float* __restrict__ weight,
                     const float* __restrict__ bias,
                     const unsigned int* __restrict__ aT_in,
                     const unsigned int* __restrict__ aB_in,
                     const float* __restrict__ w00_in,
                     const float* __restrict__ w01_in,
                     const float* __restrict__ w10_in,
                     const float* __restrict__ w11_in,
                     float* __restrict__ out) {
    __shared__ __attribute__((aligned(32))) __bf16 lA[4 * 2 * 32 * 16]; // 8 KB
    __shared__ __attribute__((aligned(32))) __bf16 lB[4 * 2 * 32 * 16]; // 8 KB
    __shared__ unsigned int lAT[KK * 64];
    __shared__ unsigned int lAB[KK * 64];
    __shared__ float lW00[KK * 64];
    __shared__ float lW01[KK * 64];
    __shared__ float lW10[KK * 64];
    __shared__ float lW11[KK * 64];

    const int tid  = threadIdx.x;
    const int lane = tid & 31;
    const int wv   = tid >> 5;
    const int mt   = wv & 3;   // 16-row block of CHO
    const int nc   = wv >> 2;  // 32-pixel half of the WG tile
    const int p0   = blockIdx.x * 64;
    const float* xb = x + (size_t)(p0 >> 15) * CHI * HW;  // batch base (uniform)

    // stage all taps' sampling state for this WG's 64 pixels
    for (int e = tid; e < KK * 64; e += 256) {
        const int tap = e >> 6;
        const int n   = e & 63;
        const int g   = tap * NPIX + p0 + n;
        lAT[e]  = aT_in[g];
        lAB[e]  = aB_in[g];
        lW00[e] = w00_in[g];
        lW01[e] = w01_in[g];
        lW10[e] = w10_in[g];
        lW11[e] = w11_in[g];
    }

    v8f c0 = {}, c1 = {};

    // per-thread staging coordinates (see fragment-order decomposition)
    const int i2   = tid & 7;            // channel pair within fragment
    const int ln   = (tid >> 3) & 31;    // fragment lane
    const int half = ln >> 4;

    for (int tap = 0; tap < KK; ++tap) {
        __syncthreads();  // protect LDS from prior-iteration readers

        // ---- stage A: weight[rows 0..31 x 64ch] for this tap, fragment order ----
#pragma unroll
        for (int it = 0; it < 8; ++it) {
            const int e    = tid + it * 256;
            const int i    = e & 15;
            const int lnA  = (e >> 4) & 31;
            const int ks   = (e >> 9) & 1;
            const int mtl  = e >> 10;
            const int m    = mtl * 16 + (lnA & 15);
            const int hA   = lnA >> 4;
            const int v    = i >> 1;
            const int pb   = i & 1;
            const int kl   = (v < 4 ? 2 * v : 16 + 2 * (v - 4)) + 8 * hA + pb;
            const int c    = ks * 32 + kl;
            lA[e] = (__bf16)weight[(m * CHI + c) * KK + tap];
        }

        // ---- stage B: gathered activations, fragment order, channel pairs --
#pragma unroll
        for (int nsub = 0; nsub < 4; ++nsub) {
            const int nloc = nsub * 16 + (ln & 15);
            const int q    = tap * 64 + nloc;
            const unsigned int pT = lAT[q];
            const unsigned int pB = lAB[q];
            const float w00 = lW00[q];
            const float w01 = lW01[q];
            const float w10 = lW10[q];
            const float w11 = lW11[q];
            const int a00 = (int)(pT & 0xFFFFu);
            const int a01 = (int)(pT >> 16);
            const int a10 = (int)(pB & 0xFFFFu);
            const int a11 = (int)(pB >> 16);
#pragma unroll
            for (int ks = 0; ks < 2; ++ks) {
                const int c    = ks * 32 + 8 * half + 2 * i2 + ((i2 >> 2) << 3);  // even channel
                const int cb   = c * HW;
                const int o00 = cb + a00, o01 = cb + a01;
                const int o10 = cb + a10, o11 = cb + a11;
                // channel c
                const float va = w00 * xb[o00] + w01 * xb[o01] +
                                 w10 * xb[o10] + w11 * xb[o11];
                // channel c+1: same addresses + HW (folds into imm offset)
                const float vb = w00 * xb[o00 + HW] + w01 * xb[o01 + HW] +
                                 w10 * xb[o10 + HW] + w11 * xb[o11 + HW];
                const int e0 = (nsub * 1024) + (ks * 512) + (ln * 16) + 2 * i2;
                *(v2bf*)&lB[e0] = (v2bf){(__bf16)va, (__bf16)vb};
            }
        }
        __syncthreads();

        // ---- MACs: 4 WMMA per tap per wave, K=64 channels ----
#pragma unroll
        for (int ks = 0; ks < 2; ++ks) {
            const v16bf a  = *(const v16bf*)&lA[(((mt << 1) | ks) * 32 + lane) << 4];
            const v16bf b0 = *(const v16bf*)&lB[((((nc * 2 + 0) << 1) | ks) * 32 + lane) << 4];
            const v16bf b1 = *(const v16bf*)&lB[((((nc * 2 + 1) << 1) | ks) * 32 + lane) << 4];
            c0 = __builtin_amdgcn_wmma_f32_16x16x32_bf16(false, a, false, b0,
                                                         (short)0, c0, false, false);
            c1 = __builtin_amdgcn_wmma_f32_16x16x32_bf16(false, a, false, b1,
                                                         (short)0, c1, false, false);
        }
    }

    // ---- epilogue: C-tile layout (ISA 7.12.2) -> NCHW stores + bias ----
    const int m0 = mt * 16 + ((lane >> 4) << 3);  // lanes 16..31 hold rows +8
    const int nl = lane & 15;
#pragma unroll
    for (int s = 0; s < 2; ++s) {
        const v8f cc = s ? c1 : c0;
        const int n = p0 + nc * 32 + s * 16 + nl;
        const int b = n >> 15;
        const int r = n & (HW - 1);
        float* ob = out + (size_t)(b * CHO) * HW + r;
#pragma unroll
        for (int j = 0; j < 8; ++j) {
            const int o = m0 + j;
            ob[(size_t)o * HW] = cc[j] + bias[o];
        }
    }
}

// ---------------------------------------------------------------------------
extern "C" void kernel_launch(void* const* d_in, const int* in_sizes, int n_in,
                              void* d_out, int out_size, void* d_ws, size_t ws_size,
                              hipStream_t stream) {
    const float* x      = (const float*)d_in[0];
    const float* w_off  = (const float*)d_in[1];
    const float* b_off  = (const float*)d_in[2];
    const float* weight = (const float*)d_in[3];
    const float* bias   = (const float*)d_in[4];
    float* out = (float*)d_out;

    const size_t plane = (size_t)KK * NPIX;
    unsigned int* aT  = (unsigned int*)d_ws;
    unsigned int* aB  = aT + plane;
    float* w00 = (float*)(aB + plane);
    float* w01 = w00 + plane;
    float* w10 = w01 + plane;
    float* w11 = w10 + plane;

    dcn_offset_kernel<<<NPIX / 256, 256, 0, stream>>>(x, w_off, b_off,
                                                      aT, aB, w00, w01, w10, w11);
    dcn_main_kernel<<<NPIX / 64, 256, 0, stream>>>(x, weight, bias,
                                                   aT, aB, w00, w01, w10, w11, out);
}
